// MultiHeadAttention_46402826666321
// MI455X (gfx1250) — matrix operand order repaired and checked by reading, not compiled, not run on hardware
//
#include <hip/hip_runtime.h>

// ---------------------------------------------------------------------------
// Multi-head attention for gfx1250 (CDNA5 / MI455X), wave32 + WMMA bf16 + TDM.
//   B=4, S=2048, D=768, H=12, dk=64
// d_out = [ output (B,S,D) f32 | weights (B,H,S,S) f32 ]
// ---------------------------------------------------------------------------

typedef __attribute__((ext_vector_type(16))) __bf16 v16bf;
typedef __attribute__((ext_vector_type(8)))  float  v8f;
typedef __attribute__((ext_vector_type(4)))  unsigned int u32x4;
typedef __attribute__((ext_vector_type(8)))  int         i32x8;
typedef __attribute__((ext_vector_type(4)))  int         i32x4;

#define D_MODEL   768
#define NUM_HEADS 12
#define DK        64
#define BATCH     4
#define SEQ       2048
#define M_TOTAL   (BATCH * SEQ)   // 8192 rows
#define KSTEPS    (D_MODEL / 32)  // 24

__device__ __forceinline__ unsigned short f32_bf16(float f) {
  unsigned int u = __float_as_uint(f);
  unsigned int r = u + 0x7FFFu + ((u >> 16) & 1u);   // round-to-nearest-even
  return (unsigned short)(r >> 16);
}

// WMMA fragment: 8 VGPRs = two 16-byte contiguous chunks per the ISA layouts.
union Frag16 {
  v16bf bf;
  uint4 q[2];
  unsigned int u[8];
};

// ---------------------------------------------------------------------------
// Tensor Data Mover: 2D tile load global -> LDS (D# per CDNA5 ISA).
// data_size = 2 bytes. All dims in elements.
// ---------------------------------------------------------------------------
#if defined(__AMDGCN__) && __has_builtin(__builtin_amdgcn_tensor_load_to_lds) && \
    __has_builtin(__builtin_amdgcn_s_wait_tensorcnt)
#define HAVE_TDM 1
#endif

#ifdef HAVE_TDM
__device__ __forceinline__ void tdm_load_2d(unsigned lds_off,
                                            unsigned long long gaddr,
                                            unsigned tensor_w, unsigned tensor_h,
                                            unsigned tile_w, unsigned tile_h,
                                            unsigned stride_w) {
  u32x4 g0;
  g0[0] = 1u;                                        // count=1, user mode
  g0[1] = lds_off;                                   // lds_addr (bytes)
  g0[2] = (unsigned)(gaddr & 0xffffffffull);         // global_addr[31:0]
  g0[3] = (unsigned)((gaddr >> 32) & 0x1ffffffull)   // global_addr[56:32]
          | (2u << 30);                              // type = 2 ("image")
  i32x8 g1;
  g1[0] = (int)(1u << 16);                           // wg_mask=0, data_size=1 (2B)
  g1[1] = (int)((tensor_w & 0xffffu) << 16);         // tensor_dim0[15:0]
  g1[2] = (int)((tensor_w >> 16) | ((tensor_h & 0xffffu) << 16));
  g1[3] = (int)((tensor_h >> 16) | (tile_w << 16));  // tile_dim0
  g1[4] = (int)(tile_h);                             // tile_dim1, tile_dim2=0
  g1[5] = (int)stride_w;                             // tensor_dim0_stride[31:0]
  g1[6] = 0;
  g1[7] = 0;
  i32x4 z4 = {0, 0, 0, 0};
#if __clang_major__ >= 23
  i32x8 z8 = {0, 0, 0, 0, 0, 0, 0, 0};
  __builtin_amdgcn_tensor_load_to_lds(g0, g1, z4, z4, z8, 0);
#else
  __builtin_amdgcn_tensor_load_to_lds(g0, g1, z4, z4, 0);
#endif
}
#endif

// A-matrix 16x32 bf16 (ISA 7.12.2): lane holds row M=lane%16; VGPR v pair base:
__device__ __forceinline__ int kA(int v, int lane) {
  return ((v & 3) * 2) + ((v >= 4) ? 16 : 0) + ((lane >= 16) ? 8 : 0);
}

// ---------------------------------------------------------------------------
// f32 -> bf16 conversion, 4 elements/thread
// ---------------------------------------------------------------------------
__global__ void f32_to_bf16_kernel(const float4* __restrict__ src,
                                   unsigned short* __restrict__ dst, int n4) {
  for (int i = blockIdx.x * blockDim.x + threadIdx.x; i < n4;
       i += gridDim.x * blockDim.x) {
    float4 f = src[i];
    uint2 p;
    p.x = (unsigned)f32_bf16(f.x) | ((unsigned)f32_bf16(f.y) << 16);
    p.y = (unsigned)f32_bf16(f.z) | ((unsigned)f32_bf16(f.w) << 16);
    *(uint2*)(dst + (size_t)i * 4) = p;
  }
}

// ---------------------------------------------------------------------------
// Fused QKV projection: 128 threads (4 waves), tile 64(M) x 16(N).
// Ping-pong LDS buffers; TDM double-buffered (1 descriptor per wave per step)
// or unrolled b128 fallback. Q scaled by 1/sqrt(dk); V stored TRANSPOSED
// (B,H,dk,S) for the AV GEMM.
// ---------------------------------------------------------------------------
__global__ __launch_bounds__(128) void qkv_gemm_kernel(
    const unsigned short* __restrict__ xb,
    const unsigned short* __restrict__ Wqb,
    const unsigned short* __restrict__ Wkb,
    const unsigned short* __restrict__ Wvb,
    const float* __restrict__ bq, const float* __restrict__ bk,
    const float* __restrict__ bv,
    unsigned short* __restrict__ Qb, unsigned short* __restrict__ Kb,
    unsigned short* __restrict__ Vt) {
  __shared__ unsigned short As[2][64 * 32];
  __shared__ unsigned short Bq[2][16 * 32];
  __shared__ unsigned short Bk[2][16 * 32];
  __shared__ unsigned short Bv[2][16 * 32];

  const int tid  = threadIdx.x;
  const int wave = __builtin_amdgcn_readfirstlane(tid >> 5);  // SGPR wave id
  const int lane = tid & 31;
  const int m0 = blockIdx.x * 64;
  const int n0 = blockIdx.y * 16;
  const int hi8  = (lane >= 16) ? 8 : 0;
  const int hi16 = (lane >= 16) ? 16 : 0;

  v8f aq = {}, ak = {}, av = {};

#ifdef HAVE_TDM
  auto issue = [&](int buf, int kk) {
    if (wave == 0) {
      tdm_load_2d((unsigned)(size_t)&As[buf][0],
                  (unsigned long long)(size_t)(xb) +
                      ((unsigned long long)m0 * D_MODEL + kk) * 2ull,
                  D_MODEL, M_TOTAL, 32, 64, D_MODEL);
    } else if (wave == 1) {
      tdm_load_2d((unsigned)(size_t)&Bq[buf][0],
                  (unsigned long long)(size_t)(Wqb) +
                      ((unsigned long long)n0 * D_MODEL + kk) * 2ull,
                  D_MODEL, D_MODEL, 32, 16, D_MODEL);
    } else if (wave == 2) {
      tdm_load_2d((unsigned)(size_t)&Bk[buf][0],
                  (unsigned long long)(size_t)(Wkb) +
                      ((unsigned long long)n0 * D_MODEL + kk) * 2ull,
                  D_MODEL, D_MODEL, 32, 16, D_MODEL);
    } else {
      tdm_load_2d((unsigned)(size_t)&Bv[buf][0],
                  (unsigned long long)(size_t)(Wvb) +
                      ((unsigned long long)n0 * D_MODEL + kk) * 2ull,
                  D_MODEL, D_MODEL, 32, 16, D_MODEL);
    }
  };
  issue(0, 0);
  for (int ks = 0; ks < KSTEPS; ++ks) {
    const int buf = ks & 1;
    __builtin_amdgcn_s_wait_tensorcnt(0);  // my tile for `buf` is in LDS
    __syncthreads();                       // everyone's tile is in LDS, and
                                           // all reads of buf^1 are finished
    if (ks + 1 < KSTEPS) issue(buf ^ 1, (ks + 1) * 32);  // overlap next DMA

    Frag16 a, fq, fk, fv;
    const int mrow = wave * 16 + (lane & 15);
    const int nrow = lane & 15;
    a.q[0]  = *(const uint4*)&As[buf][mrow * 32 + hi8];
    a.q[1]  = *(const uint4*)&As[buf][mrow * 32 + hi8 + 16];
    fq.q[0] = *(const uint4*)&Bq[buf][nrow * 32 + hi8];
    fq.q[1] = *(const uint4*)&Bq[buf][nrow * 32 + hi8 + 16];
    fk.q[0] = *(const uint4*)&Bk[buf][nrow * 32 + hi8];
    fk.q[1] = *(const uint4*)&Bk[buf][nrow * 32 + hi8 + 16];
    fv.q[0] = *(const uint4*)&Bv[buf][nrow * 32 + hi8];
    fv.q[1] = *(const uint4*)&Bv[buf][nrow * 32 + hi8 + 16];

    aq = __builtin_amdgcn_wmma_f32_16x16x32_bf16(false, a.bf, false, fq.bf,
                                                 (short)0, aq, false, false);
    ak = __builtin_amdgcn_wmma_f32_16x16x32_bf16(false, a.bf, false, fk.bf,
                                                 (short)0, ak, false, false);
    av = __builtin_amdgcn_wmma_f32_16x16x32_bf16(false, a.bf, false, fv.bf,
                                                 (short)0, av, false, false);
  }
#else
  for (int ks = 0; ks < KSTEPS; ++ks) {
    const int kk = ks * 32;
    // A tile: 64x32 bf16 = 256 uint4; 2 per thread, fully unrolled.
#pragma unroll
    for (int j = 0; j < 2; ++j) {
      int idx = tid + j * 128;
      int row = idx >> 2, q = idx & 3;
      *(uint4*)&As[0][row * 32 + q * 8] =
          *(const uint4*)(xb + (size_t)(m0 + row) * D_MODEL + kk + q * 8);
    }
    {
      int j = tid & 63;
      int n = j >> 2, q = j & 3;
      size_t wo = (size_t)(n0 + n) * D_MODEL + kk + q * 8;
      if (tid < 64)
        *(uint4*)&Bq[0][n * 32 + q * 8] = *(const uint4*)(Wqb + wo);
      else
        *(uint4*)&Bk[0][n * 32 + q * 8] = *(const uint4*)(Wkb + wo);
      if (tid < 64)
        *(uint4*)&Bv[0][n * 32 + q * 8] = *(const uint4*)(Wvb + wo);
    }
    __syncthreads();
    Frag16 a, fq, fk, fv;
    const int mrow = wave * 16 + (lane & 15);
    const int nrow = lane & 15;
    a.q[0]  = *(const uint4*)&As[0][mrow * 32 + hi8];
    a.q[1]  = *(const uint4*)&As[0][mrow * 32 + hi8 + 16];
    fq.q[0] = *(const uint4*)&Bq[0][nrow * 32 + hi8];
    fq.q[1] = *(const uint4*)&Bq[0][nrow * 32 + hi8 + 16];
    fk.q[0] = *(const uint4*)&Bk[0][nrow * 32 + hi8];
    fk.q[1] = *(const uint4*)&Bk[0][nrow * 32 + hi8 + 16];
    fv.q[0] = *(const uint4*)&Bv[0][nrow * 32 + hi8];
    fv.q[1] = *(const uint4*)&Bv[0][nrow * 32 + hi8 + 16];
    aq = __builtin_amdgcn_wmma_f32_16x16x32_bf16(false, a.bf, false, fq.bf,
                                                 (short)0, aq, false, false);
    ak = __builtin_amdgcn_wmma_f32_16x16x32_bf16(false, a.bf, false, fk.bf,
                                                 (short)0, ak, false, false);
    av = __builtin_amdgcn_wmma_f32_16x16x32_bf16(false, a.bf, false, fv.bf,
                                                 (short)0, av, false, false);
    __syncthreads();
  }
#endif

  // Epilogue: f32 bias add; Q/K -> (B,H,S,dk) bf16, V -> (B,H,dk,S) bf16.
  const int n = n0 + (lane & 15);
  const int h = n >> 6, d = n & 63;
  const float bqv = bq[n], bkv = bk[n], bvv = bv[n];
#pragma unroll
  for (int r = 0; r < 8; ++r) {
    int m = m0 + wave * 16 + r + hi8;
    int bidx = m >> 11;
    int s = m & (SEQ - 1);
    size_t o = ((size_t)(bidx * NUM_HEADS + h) * SEQ + s) * DK + d;
    Qb[o] = f32_bf16((aq[r] + bqv) * 0.125f);
    Kb[o] = f32_bf16(ak[r] + bkv);
    size_t ot = ((size_t)(bidx * NUM_HEADS + h) * DK + d) * SEQ + s;
    Vt[ot] = f32_bf16(av[r] + bvv);
  }
}

// ---------------------------------------------------------------------------
// Attention: workgroup = (b, h, 16-query tile), 256 threads = 8 waves.
// Scores (16x2048 f32 = 128KB) in LDS, softmax in LDS, weights written once,
// context = weights @ V^T via WMMA with ds_add_f32 cross-wave reduction.
// ---------------------------------------------------------------------------
__global__ __launch_bounds__(256) void attention_kernel(
    const unsigned short* __restrict__ Qb,
    const unsigned short* __restrict__ Kb,
    const unsigned short* __restrict__ Vt,
    unsigned short* __restrict__ Cb,   // merged context bf16 (B,S,D)
    float* __restrict__ Wout) {        // attention weights (B,H,S,S) f32
  extern __shared__ float smem[];
  float* sS   = smem;              // 16*2048
  float* sRed = sS + 16 * SEQ;     // 256
  float* sMax = sRed + 256;        // 16
  float* sSum = sMax + 16;         // 16
  float* sCtx = sSum + 16;         // 16*64

  const int tid  = threadIdx.x;
  const int wave = __builtin_amdgcn_readfirstlane(tid >> 5);
  const int lane = tid & 31;
  const int hi8  = (lane >= 16) ? 8 : 0;
  const int hi16 = (lane >= 16) ? 16 : 0;

  const int blk = blockIdx.x;
  const int qt = blk & 127;
  const int h  = (blk >> 7) % NUM_HEADS;
  const int b  = blk / (128 * NUM_HEADS);
  const int q0 = qt * 16;
  const size_t bh  = (size_t)(b * NUM_HEADS + h) * SEQ;   // Q/K row base
  const size_t bhd = (size_t)(b * NUM_HEADS + h) * DK;    // V^T row base

  // ---- Phase 1: scores = Q * K^T ----
  Frag16 qf0, qf1;
  {
    const unsigned short* qrow = Qb + (bh + q0 + (lane & 15)) * DK;
    qf0.q[0] = *(const uint4*)(qrow + hi8);
    qf0.q[1] = *(const uint4*)(qrow + hi8 + 16);
    qf1.q[0] = *(const uint4*)(qrow + 32 + hi8);
    qf1.q[1] = *(const uint4*)(qrow + 32 + hi8 + 16);
  }
  for (int t = 0; t < 16; ++t) {
    const int k0 = wave * 256 + t * 16;
    const int n = lane & 15;
    const unsigned short* krow = Kb + (bh + k0 + n) * DK;
    if (t + 1 < 16)  // get next 2KB K tile moving (global_prefetch_b8)
      __builtin_prefetch(krow + 16 * DK, 0, 0);
    v8f acc = {};
    Frag16 kf;
    kf.q[0] = *(const uint4*)(krow + hi8);
    kf.q[1] = *(const uint4*)(krow + hi8 + 16);
    acc = __builtin_amdgcn_wmma_f32_16x16x32_bf16(false, qf0.bf, false, kf.bf,
                                                  (short)0, acc, false, false);
    kf.q[0] = *(const uint4*)(krow + 32 + hi8);
    kf.q[1] = *(const uint4*)(krow + 32 + hi8 + 16);
    acc = __builtin_amdgcn_wmma_f32_16x16x32_bf16(false, qf1.bf, false, kf.bf,
                                                  (short)0, acc, false, false);
#pragma unroll
    for (int r = 0; r < 8; ++r)
      sS[(r + hi8) * SEQ + k0 + n] = acc[r];
  }
  __syncthreads();

  // ---- Phase 2: row max & exp-sum (16 rows x 16 threads) ----
  const int row = tid >> 4, l16 = tid & 15;
  {
    float m = -3.0e38f;
    for (int j = l16; j < SEQ; j += 16) m = fmaxf(m, sS[row * SEQ + j]);
    sRed[row * 16 + l16] = m;
    __syncthreads();
    if (l16 == 0) {
      float mm = sRed[row * 16];
#pragma unroll
      for (int i = 1; i < 16; ++i) mm = fmaxf(mm, sRed[row * 16 + i]);
      sMax[row] = mm;
    }
    __syncthreads();
    const float rmax = sMax[row];
    float s = 0.f;
    for (int j = l16; j < SEQ; j += 16) s += __expf(sS[row * SEQ + j] - rmax);
    sRed[row * 16 + l16] = s;
    __syncthreads();
    if (l16 == 0) {
      float ss = 0.f;
#pragma unroll
      for (int i = 0; i < 16; ++i) ss += sRed[row * 16 + i];
      sSum[row] = ss;
    }
    __syncthreads();

    // ---- Phase 3: normalize in LDS + single HBM write of weights ----
    const float inv = 1.0f / sSum[row];
    float* wrow = Wout + (bh + q0 + row) * SEQ;
    for (int j = l16; j < SEQ; j += 16) {
      float w = __expf(sS[row * SEQ + j] - rmax) * inv;
      sS[row * SEQ + j] = w;
      wrow[j] = w;
    }
  }
  for (int i = tid; i < 16 * 64; i += 256) sCtx[i] = 0.f;
  __syncthreads();

  // ---- Phase 4: context = weights @ V, K-split across 8 waves ----
  {
    v8f acc0 = {}, acc1 = {}, acc2 = {}, acc3 = {};
    const int kbase = wave * 256;
    const int m = lane & 15;
    for (int step = 0; step < 8; ++step) {
      const int k0 = kbase + step * 32;
      Frag16 af;
#pragma unroll
      for (int v = 0; v < 8; ++v) {   // pairs are contiguous f32 -> ds_load_b64
        int kk = k0 + kA(v, lane);
        float2 p = *(const float2*)&sS[m * SEQ + kk];
        af.u[v] = (unsigned)f32_bf16(p.x) | ((unsigned)f32_bf16(p.y) << 16);
      }
#pragma unroll
      for (int dt = 0; dt < 4; ++dt) {
        Frag16 bfr;
        const int n = dt * 16 + (lane & 15);
        const unsigned short* vrow = Vt + (bhd + n) * SEQ + k0;
        if (step + 1 < 8) __builtin_prefetch(vrow + 32, 0, 0);
        bfr.q[0] = *(const uint4*)(vrow + hi8);
        bfr.q[1] = *(const uint4*)(vrow + hi8 + 16);
        v8f* accp = dt == 0 ? &acc0 : dt == 1 ? &acc1 : dt == 2 ? &acc2 : &acc3;
        *accp = __builtin_amdgcn_wmma_f32_16x16x32_bf16(
            false, af.bf, false, bfr.bf, (short)0, *accp, false, false);
      }
    }
    const int N = lane & 15;
#pragma unroll
    for (int r = 0; r < 8; ++r) {
      atomicAdd(&sCtx[(r + hi8) * 64 + 0 * 16 + N], acc0[r]);
      atomicAdd(&sCtx[(r + hi8) * 64 + 1 * 16 + N], acc1[r]);
      atomicAdd(&sCtx[(r + hi8) * 64 + 2 * 16 + N], acc2[r]);
      atomicAdd(&sCtx[(r + hi8) * 64 + 3 * 16 + N], acc3[r]);
    }
  }
  __syncthreads();

  // ---- Phase 5: merged context (B,S,D) bf16 ----
  for (int i = tid; i < 16 * 64; i += 256) {
    int m = i >> 6, d = i & 63;
    Cb[((size_t)(b * SEQ + q0 + m)) * D_MODEL + h * DK + d] = f32_bf16(sCtx[i]);
  }
}

// ---------------------------------------------------------------------------
// Output projection: out = merged @ Wo^T + bo (f32 straight to d_out)
// ---------------------------------------------------------------------------
__global__ __launch_bounds__(128) void out_gemm_kernel(
    const unsigned short* __restrict__ Cb,
    const unsigned short* __restrict__ Wob,
    const float* __restrict__ bo, float* __restrict__ out) {
  __shared__ unsigned short As[2][64 * 32];
  __shared__ unsigned short Bo[2][16 * 32];

  const int tid  = threadIdx.x;
  const int wave = __builtin_amdgcn_readfirstlane(tid >> 5);
  const int lane = tid & 31;
  const int m0 = blockIdx.x * 64;
  const int n0 = blockIdx.y * 16;
  const int hi8  = (lane >= 16) ? 8 : 0;
  const int hi16 = (lane >= 16) ? 16 : 0;

  v8f acc = {};

#ifdef HAVE_TDM
  auto issue = [&](int buf, int kk) {
    if (wave == 0) {
      tdm_load_2d((unsigned)(size_t)&As[buf][0],
                  (unsigned long long)(size_t)(Cb) +
                      ((unsigned long long)m0 * D_MODEL + kk) * 2ull,
                  D_MODEL, M_TOTAL, 32, 64, D_MODEL);
    } else if (wave == 1) {
      tdm_load_2d((unsigned)(size_t)&Bo[buf][0],
                  (unsigned long long)(size_t)(Wob) +
                      ((unsigned long long)n0 * D_MODEL + kk) * 2ull,
                  D_MODEL, D_MODEL, 32, 16, D_MODEL);
    }
  };
  issue(0, 0);
  for (int ks = 0; ks < KSTEPS; ++ks) {
    const int buf = ks & 1;
    __builtin_amdgcn_s_wait_tensorcnt(0);
    __syncthreads();
    if (ks + 1 < KSTEPS) issue(buf ^ 1, (ks + 1) * 32);

    Frag16 a, bfr;
    const int mrow = wave * 16 + (lane & 15);
    const int nrow = lane & 15;
    a.q[0]   = *(const uint4*)&As[buf][mrow * 32 + hi8];
    a.q[1]   = *(const uint4*)&As[buf][mrow * 32 + hi8 + 16];
    bfr.q[0] = *(const uint4*)&Bo[buf][nrow * 32 + hi8];
    bfr.q[1] = *(const uint4*)&Bo[buf][nrow * 32 + hi8 + 16];
    acc = __builtin_amdgcn_wmma_f32_16x16x32_bf16(false, a.bf, false, bfr.bf,
                                                  (short)0, acc, false, false);
  }
#else
  for (int ks = 0; ks < KSTEPS; ++ks) {
    const int kk = ks * 32;
#pragma unroll
    for (int j = 0; j < 2; ++j) {
      int idx = tid + j * 128;
      int row = idx >> 2, q = idx & 3;
      *(uint4*)&As[0][row * 32 + q * 8] =
          *(const uint4*)(Cb + (size_t)(m0 + row) * D_MODEL + kk + q * 8);
    }
    if (tid < 64) {
      int n = tid >> 2, q = tid & 3;
      *(uint4*)&Bo[0][n * 32 + q * 8] =
          *(const uint4*)(Wob + (size_t)(n0 + n) * D_MODEL + kk + q * 8);
    }
    __syncthreads();
    Frag16 a, bfr;
    const int mrow = wave * 16 + (lane & 15);
    const int nrow = lane & 15;
    a.q[0]   = *(const uint4*)&As[0][mrow * 32 + hi8];
    a.q[1]   = *(const uint4*)&As[0][mrow * 32 + hi8 + 16];
    bfr.q[0] = *(const uint4*)&Bo[0][nrow * 32 + hi8];
    bfr.q[1] = *(const uint4*)&Bo[0][nrow * 32 + hi8 + 16];
    acc = __builtin_amdgcn_wmma_f32_16x16x32_bf16(false, a.bf, false, bfr.bf,
                                                  (short)0, acc, false, false);
    __syncthreads();
  }
#endif

  const int n = n0 + (lane & 15);
  const float bov = bo[n];
#pragma unroll
  for (int r = 0; r < 8; ++r) {
    int m = m0 + wave * 16 + r + hi8;
    out[(size_t)m * D_MODEL + n] = acc[r] + bov;
  }
}

// ---------------------------------------------------------------------------
// Host launcher
// ---------------------------------------------------------------------------
extern "C" void kernel_launch(void* const* d_in, const int* in_sizes, int n_in,
                              void* d_out, int out_size, void* d_ws,
                              size_t ws_size, hipStream_t stream) {
  const float* x  = (const float*)d_in[0];
  const float* Wq = (const float*)d_in[1];
  const float* bq = (const float*)d_in[2];
  const float* Wk = (const float*)d_in[3];
  const float* bk = (const float*)d_in[4];
  const float* Wv = (const float*)d_in[5];
  const float* bv = (const float*)d_in[6];
  const float* Wo = (const float*)d_in[7];
  const float* bo = (const float*)d_in[8];

  char* ws = (char*)d_ws;
  size_t off = 0;
  auto alloc = [&](size_t bytes) -> char* {
    char* p = ws + off;
    off += (bytes + 255) & ~(size_t)255;
    return p;
  };
  const size_t nX = (size_t)M_TOTAL * D_MODEL;  // 6.29M elems
  const size_t nW = (size_t)D_MODEL * D_MODEL;  // 0.59M elems

  unsigned short* xb  = (unsigned short*)alloc(nX * 2);
  unsigned short* Wqb = (unsigned short*)alloc(nW * 2);
  unsigned short* Wkb = (unsigned short*)alloc(nW * 2);
  unsigned short* Wvb = (unsigned short*)alloc(nW * 2);
  unsigned short* Wob = (unsigned short*)alloc(nW * 2);
  unsigned short* Qb  = (unsigned short*)alloc(nX * 2);  // (B,H,S,dk)
  unsigned short* Kb  = (unsigned short*)alloc(nX * 2);  // (B,H,S,dk)
  unsigned short* Vt  = (unsigned short*)alloc(nX * 2);  // (B,H,dk,S)
  unsigned short* Cb  = (unsigned short*)alloc(nX * 2);  // merged context

  float* out     = (float*)d_out;  // (B,S,D)
  float* weights = out + nX;       // (B,H,S,S)

  f32_to_bf16_kernel<<<1024, 256, 0, stream>>>((const float4*)x, xb,
                                               (int)(nX / 4));
  f32_to_bf16_kernel<<<256, 256, 0, stream>>>((const float4*)Wq, Wqb,
                                              (int)(nW / 4));
  f32_to_bf16_kernel<<<256, 256, 0, stream>>>((const float4*)Wk, Wkb,
                                              (int)(nW / 4));
  f32_to_bf16_kernel<<<256, 256, 0, stream>>>((const float4*)Wv, Wvb,
                                              (int)(nW / 4));
  f32_to_bf16_kernel<<<256, 256, 0, stream>>>((const float4*)Wo, Wob,
                                              (int)(nW / 4));

  dim3 gGemm(M_TOTAL / 64, D_MODEL / 16);
  qkv_gemm_kernel<<<gGemm, 128, 0, stream>>>(xb, Wqb, Wkb, Wvb, bq, bk, bv,
                                             Qb, Kb, Vt);

  const size_t shbytes =
      (size_t)(16 * SEQ + 256 + 16 + 16 + 16 * 64) * sizeof(float);  // ~136KB
  attention_kernel<<<BATCH * NUM_HEADS * (SEQ / 16), 256, shbytes, stream>>>(
      Qb, Kb, Vt, Cb, weights);

  out_gemm_kernel<<<gGemm, 128, 0, stream>>>(Cb, Wob, bo, out);
}
